// PrefixViTBackbone_549755813928
// MI455X (gfx1250) — matrix operand order repaired and checked by reading, not compiled, not run on hardware
//
#include <hip/hip_runtime.h>
#include <hip/hip_bf16.h>

// ---------------- model constants ----------------
#define B_      32
#define IMG_    224
#define P_      16
#define D_      768
#define H_      12
#define HD_     64
#define DEPTH_  12
#define MLPD_   3072
#define N_      197
#define NP_     196
#define TASKS_  10
#define PL_     5
#define NPL_    5
#define M_ROWS  (B_ * N_)     // 6304
#define M_PATCH (B_ * NP_)    // 6272
#define SCALE_  0.125f        // HD^-0.5

typedef __attribute__((ext_vector_type(16))) _Float16 v16h;
typedef __attribute__((ext_vector_type(8)))  _Float16 v8h;
typedef __attribute__((ext_vector_type(8)))  float    v8f;

// ---------------------------------------------------------------------
// CDNA5 async global->LDS copy (ASYNCcnt path), 16B per lane.
// lds_off = byte offset within the wave's LDS allocation (low 32 bits of
// a generic shared pointer, per the ISA aperture rule).
// ---------------------------------------------------------------------
__device__ __forceinline__ void async_copy_b128(uint32_t lds_off, const void* gptr)
{
    asm volatile("global_load_async_to_lds_b128 %0, %1, off"
                 :: "v"(lds_off), "v"(gptr)
                 : "memory");
}
__device__ __forceinline__ void wait_asynccnt0()
{
    asm volatile("s_wait_asynccnt 0" ::: "memory");
}

// =====================================================================
// WMMA GEMM, all-f16 operands:  out[m,n] = epi( sum_k A[m,k] * Bw[n,k] )
// A:  f16 row-major MxK (activations). Bw: f16 row-major NxK (weights,
// pre-converted). Epilogue: +bias, optional exact GELU, optional +resid.
// Tile: BM=64 BN=128 BK=64; 256 thr = 8 waves (2x4); wave = 32x32 out
// (2x2 WMMA tiles). Double-buffered LDS staged with async b128 copies.
// =====================================================================
__global__ __launch_bounds__(256)
void gemm_f16_wmma(const _Float16* __restrict__ A, const _Float16* __restrict__ Bw,
                   const float* __restrict__ bias, const float* __restrict__ resid,
                   float* __restrict__ outF, _Float16* __restrict__ outH,
                   int M, int Nc, int K, int gelu)
{
    __shared__ __align__(16) _Float16 sA[2][64 * 64];    // 16 KB
    __shared__ __align__(16) _Float16 sB[2][128 * 64];   // 32 KB

    const int tid    = threadIdx.x;
    const int m0     = blockIdx.x * 64;
    const int n0     = blockIdx.y * 128;
    const int lane   = tid & 31;
    const int wv     = tid >> 5;            // wave 0..7
    const int waveM  = (wv >> 2) * 32;      // 0 / 32
    const int waveN  = (wv & 3) * 32;       // 0 / 32 / 64 / 96
    const int halfid = lane >> 4;
    const int l15    = lane & 15;

    const uint32_t ldsA0 = (uint32_t)(uintptr_t)(&sA[0][0]);
    const uint32_t ldsB0 = (uint32_t)(uintptr_t)(&sB[0][0]);

    v8f acc00 = {}, acc01 = {}, acc10 = {}, acc11 = {};

    // ---------- tile staging (async copies; zero-fill OOB rows) ----------
    auto stage = [&](int buf, int k0) {
        // A: 64 rows x 128B = 512 chunks of 16B, 2 per thread
        #pragma unroll
        for (int c = 0; c < 2; ++c) {
            int chunk = tid + c * 256;
            int row   = chunk >> 3;
            int coff  = (chunk & 7) * 16;
            uint32_t lo = ldsA0 + (uint32_t)buf * (64 * 128) + row * 128 + coff;
            int gr = m0 + row;
            if (gr < M) {
                const char* g = (const char*)(A + (size_t)gr * K + k0) + coff;
                async_copy_b128(lo, g);
            } else {
                v8h z = {};
                *(v8h*)((char*)&sA[buf][0] + row * 128 + coff) = z;
            }
        }
        // B: 128 rows x 128B = 1024 chunks of 16B, 4 per thread
        #pragma unroll
        for (int c = 0; c < 4; ++c) {
            int chunk = tid + c * 256;
            int row   = chunk >> 3;
            int coff  = (chunk & 7) * 16;
            uint32_t lo = ldsB0 + (uint32_t)buf * (128 * 128) + row * 128 + coff;
            const char* g = (const char*)(Bw + (size_t)(n0 + row) * K + k0) + coff;
            async_copy_b128(lo, g);
            if (c == 0 && k0 + 128 < K)
                __builtin_prefetch(g + 256, 0, 1);   // global_prefetch_b8, 2 tiles ahead
        }
    };

    stage(0, 0);

    int buf = 0;
    for (int k0 = 0; k0 < K; k0 += 64, buf ^= 1) {
        wait_asynccnt0();        // this wave's async stores have landed
        __syncthreads();         // everyone's stores visible; prev compute done
        if (k0 + 64 < K) stage(buf ^ 1, k0 + 64);

        const _Float16* pA = &sA[buf][0];
        const _Float16* pB = &sB[buf][0];

        #pragma unroll
        for (int ks = 0; ks < 64; ks += 32) {
            v16h a0, a1, b0, b1;
            #pragma unroll
            for (int e = 0; e < 16; ++e) {
                int i = e >> 1, r = e & 1;
                // A fragment (16x32): M = lane&15, K split by lane-half
                int ka = 2 * i + r + halfid * 16;
                a0[e] = pA[(waveM +  0 + l15) * 64 + ks + ka];
                a1[e] = pA[(waveM + 16 + l15) * 64 + ks + ka];
                // B fragment (32x16): N = lane&15, K split by lane-half
                int kb = 2 * i + r + halfid * 16;
                b0[e] = pB[(waveN +  0 + l15) * 64 + ks + kb];
                b1[e] = pB[(waveN + 16 + l15) * 64 + ks + kb];
            }
            acc00 = __builtin_amdgcn_wmma_f32_16x16x32_f16(false, a0, false, b0,
                                                           (short)0, acc00, false, false);
            acc01 = __builtin_amdgcn_wmma_f32_16x16x32_f16(false, a0, false, b1,
                                                           (short)0, acc01, false, false);
            acc10 = __builtin_amdgcn_wmma_f32_16x16x32_f16(false, a1, false, b0,
                                                           (short)0, acc10, false, false);
            acc11 = __builtin_amdgcn_wmma_f32_16x16x32_f16(false, a1, false, b1,
                                                           (short)0, acc11, false, false);
        }
    }

    // ---- epilogue: C/D layout M = v + 8*halfid, N = lane&15 ----
    const int col0 = n0 + waveN + l15;
    const int col1 = col0 + 16;
    const float bv0 = bias ? bias[col0] : 0.0f;
    const float bv1 = bias ? bias[col1] : 0.0f;
    #pragma unroll
    for (int v = 0; v < 8; ++v) {
        int r0 = m0 + waveM + 8 * halfid + v;
        int r1 = r0 + 16;
        float vals[4] = { acc00[v] + bv0, acc01[v] + bv1,
                          acc10[v] + bv0, acc11[v] + bv1 };
        int   rows[4] = { r0, r0, r1, r1 };
        int   cols[4] = { col0, col1, col0, col1 };
        #pragma unroll
        for (int t = 0; t < 4; ++t) {
            if (rows[t] < M) {
                float val = vals[t];
                if (gelu)  val = 0.5f * val * (1.0f + erff(val * 0.70710678118f));
                if (resid) val += resid[(size_t)rows[t] * Nc + cols[t]];
                if (outF)  outF[(size_t)rows[t] * Nc + cols[t]] = val;
                if (outH)  outH[(size_t)rows[t] * Nc + cols[t]] = (_Float16)val;
            }
        }
    }
}

// =====================================================================
// f32 -> f16 weight conversion (per-GEMM staging; n multiple of 4)
// =====================================================================
__global__ __launch_bounds__(256)
void cvt_w_kernel(const float* __restrict__ src, _Float16* __restrict__ dst, int n)
{
    int i = (blockIdx.x * 256 + threadIdx.x) * 4;
    if (i >= n) return;
    float4 v = *(const float4*)(src + i);
    dst[i + 0] = (_Float16)v.x;
    dst[i + 1] = (_Float16)v.y;
    dst[i + 2] = (_Float16)v.z;
    dst[i + 3] = (_Float16)v.w;
}

// =====================================================================
// LayerNorm over D=768. One block per output row (256 thr, 3 elem/thr).
// =====================================================================
__global__ __launch_bounds__(256)
void layernorm_kernel(const float* __restrict__ x, const float* __restrict__ w,
                      const float* __restrict__ b, _Float16* __restrict__ outH,
                      float* __restrict__ outF, int row_mul)
{
    __shared__ float red[256];
    const int tid = threadIdx.x;
    const size_t row = (size_t)blockIdx.x * row_mul;
    const float* xr = x + row * D_;

    float s = 0.f;
    #pragma unroll
    for (int j = 0; j < 3; ++j) s += xr[tid + 256 * j];
    red[tid] = s;
    __syncthreads();
    for (int o = 128; o > 0; o >>= 1) {
        if (tid < o) red[tid] += red[tid + o];
        __syncthreads();
    }
    const float mean = red[0] * (1.0f / D_);
    __syncthreads();

    float s2 = 0.f;
    #pragma unroll
    for (int j = 0; j < 3; ++j) {
        float d = xr[tid + 256 * j] - mean;
        s2 += d * d;
    }
    red[tid] = s2;
    __syncthreads();
    for (int o = 128; o > 0; o >>= 1) {
        if (tid < o) red[tid] += red[tid + o];
        __syncthreads();
    }
    const float rstd = rsqrtf(red[0] * (1.0f / D_) + 1e-6f);

    #pragma unroll
    for (int j = 0; j < 3; ++j) {
        int c = tid + 256 * j;
        float v = (xr[c] - mean) * rstd * w[c] + b[c];
        if (outH) outH[(size_t)blockIdx.x * D_ + c] = (_Float16)v;
        if (outF) outF[(size_t)blockIdx.x * D_ + c] = v;
    }
}

// =====================================================================
// Prefix attention: block = (batch, head); K/V staged in LDS (f16);
// per-wave query rows; shfl softmax; fused prefix sigmoid gate.
// =====================================================================
__device__ __forceinline__ float wave_max32(float v) {
    #pragma unroll
    for (int m = 16; m > 0; m >>= 1) v = fmaxf(v, __shfl_xor(v, m, 32));
    return v;
}
__device__ __forceinline__ float wave_sum32(float v) {
    #pragma unroll
    for (int m = 16; m > 0; m >>= 1) v += __shfl_xor(v, m, 32);
    return v;
}

__global__ __launch_bounds__(256)
void attn_kernel(const float* __restrict__ qkv,
                 const float* __restrict__ prefix_k, const float* __restrict__ prefix_v,
                 const int* __restrict__ task_id, const float* __restrict__ act_scale,
                 _Float16* __restrict__ outH, int layer, int nkv)
{
    __shared__ _Float16 sK[(PL_ + N_) * HD_];
    __shared__ _Float16 sV[(PL_ + N_) * HD_];
    __shared__ float    sQ[8 * HD_];
    __shared__ float    sL[8 * 224];

    const int h    = blockIdx.x;
    const int b    = blockIdx.y;
    const int tid  = threadIdx.x;
    const int lane = tid & 31;
    const int wv   = tid >> 5;
    const int npre = (nkv > N_) ? PL_ : 0;

    int t = task_id[b];
    t = t < 0 ? 0 : (t > TASKS_ - 1 ? TASKS_ - 1 : t);

    for (int idx = tid; idx < nkv * HD_; idx += 256) {
        int kk = idx / HD_, d = idx & (HD_ - 1);
        float kvK, kvV;
        if (kk < npre) {
            size_t po = ((((size_t)layer * TASKS_ + t) * PL_ + kk) * H_ + h) * HD_ + d;
            kvK = prefix_k[po];
            kvV = prefix_v[po];
        } else {
            int n = kk - npre;
            size_t qo = ((size_t)(b * N_ + n)) * (3 * D_) + h * HD_ + d;
            kvK = qkv[qo + D_];
            kvV = qkv[qo + 2 * D_];
        }
        sK[idx] = (_Float16)kvK;
        sV[idx] = (_Float16)kvV;
    }
    __syncthreads();

    const float s0 = (npre > 0) ? act_scale[layer * 2 + 0] : 0.f;
    const float s1 = (npre > 0) ? act_scale[layer * 2 + 1] : 0.f;

    for (int q = wv; q < N_; q += 8) {
        const size_t qoff = ((size_t)(b * N_ + q)) * (3 * D_) + h * HD_;
        sQ[wv * HD_ + lane]      = qkv[qoff + lane];
        sQ[wv * HD_ + lane + 32] = qkv[qoff + lane + 32];

        for (int kk = lane; kk < nkv; kk += 32) {
            float dot = 0.f;
            #pragma unroll 8
            for (int d = 0; d < HD_; ++d)
                dot += sQ[wv * HD_ + d] * (float)sK[kk * HD_ + d];
            float lg = dot * SCALE_;
            if (kk < npre) {
                float sig = 1.0f / (1.0f + expf(-lg * s0));
                lg = lg + sig * s1;
            }
            sL[wv * 224 + kk] = lg;
        }
        float mx = -3.0e38f;
        for (int kk = lane; kk < nkv; kk += 32) mx = fmaxf(mx, sL[wv * 224 + kk]);
        mx = wave_max32(mx);
        float sm = 0.f;
        for (int kk = lane; kk < nkv; kk += 32) {
            float e = expf(sL[wv * 224 + kk] - mx);
            sL[wv * 224 + kk] = e;
            sm += e;
        }
        sm = wave_sum32(sm);
        const float inv = 1.0f / sm;

        #pragma unroll
        for (int half = 0; half < 2; ++half) {
            int d = lane + 32 * half;
            float acc = 0.f;
            for (int kk = 0; kk < nkv; ++kk)
                acc += sL[wv * 224 + kk] * (float)sV[kk * HD_ + d];
            outH[((size_t)(b * N_ + q)) * D_ + h * HD_ + d] = (_Float16)(acc * inv);
        }
    }
}

// =====================================================================
// im2col for the stride-16 patch conv
// =====================================================================
__global__ __launch_bounds__(256)
void im2col_kernel(const float* __restrict__ x, _Float16* __restrict__ out)
{
    int idx = blockIdx.x * 256 + threadIdx.x;
    if (idx >= M_PATCH * D_) return;
    int row = idx / D_;
    int k   = idx - row * D_;
    int b = row / NP_, pi = row - b * NP_;
    int c = k >> 8, rem = k & 255, py = rem >> 4, px = rem & 15;
    int iy = (pi / 14) * P_ + py;
    int ix = (pi % 14) * P_ + px;
    out[idx] = (_Float16)x[(((size_t)b * 3 + c) * IMG_ + iy) * IMG_ + ix];
}

// =====================================================================
// Assemble h: cls token + patch embeddings + bias + positional embed
// =====================================================================
__global__ __launch_bounds__(256)
void assemble_kernel(const float* __restrict__ ptmp, const float* __restrict__ patch_b,
                     const float* __restrict__ cls, const float* __restrict__ pos,
                     float* __restrict__ hbuf)
{
    int idx = blockIdx.x * 256 + threadIdx.x;
    if (idx >= M_ROWS * D_) return;
    int row = idx / D_;
    int col = idx - row * D_;
    int b = row / N_, n = row - b * N_;
    float v;
    if (n == 0)
        v = cls[col] + pos[col];
    else
        v = ptmp[((size_t)b * NP_ + (n - 1)) * D_ + col] + patch_b[col]
            + pos[(size_t)n * D_ + col];
    hbuf[idx] = v;
}

// =====================================================================
// host-side orchestration
// =====================================================================
extern "C" void kernel_launch(void* const* d_in, const int* in_sizes, int n_in,
                              void* d_out, int out_size, void* d_ws, size_t ws_size,
                              hipStream_t stream)
{
    (void)in_sizes; (void)n_in; (void)out_size; (void)ws_size;

    const float* x        = (const float*)d_in[0];
    const int*   task_id  = (const int*)  d_in[1];
    const float* patch_w  = (const float*)d_in[2];
    const float* patch_b  = (const float*)d_in[3];
    const float* cls_tok  = (const float*)d_in[4];
    const float* pos_emb  = (const float*)d_in[5];
    const float* ln1_w    = (const float*)d_in[6];
    const float* ln1_b    = (const float*)d_in[7];
    const float* qkv_w    = (const float*)d_in[8];
    const float* qkv_b    = (const float*)d_in[9];
    const float* proj_w   = (const float*)d_in[10];
    const float* proj_b   = (const float*)d_in[11];
    const float* ln2_w    = (const float*)d_in[12];
    const float* ln2_b    = (const float*)d_in[13];
    const float* fc1_w    = (const float*)d_in[14];
    const float* fc1_b    = (const float*)d_in[15];
    const float* fc2_w    = (const float*)d_in[16];
    const float* fc2_b    = (const float*)d_in[17];
    const float* norm_w   = (const float*)d_in[18];
    const float* norm_b   = (const float*)d_in[19];
    const float* prefix_k = (const float*)d_in[20];
    const float* prefix_v = (const float*)d_in[21];
    const float* act_sc   = (const float*)d_in[22];

    // ---- workspace layout (all 16B-aligned) ----
    char* ws = (char*)d_ws;
    size_t off = 0;
    float*    hbuf = (float*)(ws + off);    off += (size_t)M_ROWS * D_ * sizeof(float);
    float*    qkvb = (float*)(ws + off);    off += (size_t)M_ROWS * 3 * D_ * sizeof(float);
    _Float16* xnh  = (_Float16*)(ws + off); off += (size_t)M_ROWS * D_ * 2;
    _Float16* atnh = (_Float16*)(ws + off); off += (size_t)M_ROWS * D_ * 2;
    _Float16* mlph = (_Float16*)(ws + off); off += (size_t)M_ROWS * MLPD_ * 2;
    _Float16* wh   = (_Float16*)(ws + off); off += (size_t)MLPD_ * D_ * 2;   // weight staging
    float*    ptmp = qkvb;                  // patch-GEMM temp aliases qkv buffer
    _Float16* aim  = mlph;                  // im2col temp aliases MLP buffer

    const dim3 blk(256);
    auto cvt = [&](const float* src, int n) {
        cvt_w_kernel<<<dim3((n / 4 + 255) / 256), blk, 0, stream>>>(src, wh, n);
    };

    // ---- patch embedding: im2col + WMMA GEMM (6272 x 768 x 768) ----
    im2col_kernel<<<dim3((M_PATCH * D_ + 255) / 256), blk, 0, stream>>>(x, aim);
    cvt(patch_w, D_ * D_);
    gemm_f16_wmma<<<dim3(M_PATCH / 64, D_ / 128), blk, 0, stream>>>(
        aim, wh, nullptr, nullptr, ptmp, nullptr, M_PATCH, D_, D_, 0);
    assemble_kernel<<<dim3((M_ROWS * D_ + 255) / 256), blk, 0, stream>>>(
        ptmp, patch_b, cls_tok, pos_emb, hbuf);

    const int gM = (M_ROWS + 63) / 64;   // 99

    for (int i = 0; i < DEPTH_; ++i) {
        // LN1 -> f16 operand
        layernorm_kernel<<<dim3(M_ROWS), blk, 0, stream>>>(
            hbuf, ln1_w + i * D_, ln1_b + i * D_, xnh, nullptr, 1);
        // QKV GEMM (6304 x 2304 x 768)
        cvt(qkv_w + (size_t)i * 3 * D_ * D_, 3 * D_ * D_);
        gemm_f16_wmma<<<dim3(gM, (3 * D_) / 128), blk, 0, stream>>>(
            xnh, wh, qkv_b + (size_t)i * 3 * D_,
            nullptr, qkvb, nullptr, M_ROWS, 3 * D_, D_, 0);
        // attention (prefix on first NPL layers)
        int nkv = (i < NPL_) ? (N_ + PL_) : N_;
        attn_kernel<<<dim3(H_, B_), blk, 0, stream>>>(
            qkvb, prefix_k, prefix_v, task_id, act_sc, atnh, i, nkv);
        // proj GEMM + residual (6304 x 768 x 768)
        cvt(proj_w + (size_t)i * D_ * D_, D_ * D_);
        gemm_f16_wmma<<<dim3(gM, D_ / 128), blk, 0, stream>>>(
            atnh, wh, proj_b + (size_t)i * D_,
            hbuf, hbuf, nullptr, M_ROWS, D_, D_, 0);
        // LN2 -> f16 operand
        layernorm_kernel<<<dim3(M_ROWS), blk, 0, stream>>>(
            hbuf, ln2_w + i * D_, ln2_b + i * D_, xnh, nullptr, 1);
        // FC1 GEMM + exact GELU (6304 x 3072 x 768), f16 out
        cvt(fc1_w + (size_t)i * MLPD_ * D_, MLPD_ * D_);
        gemm_f16_wmma<<<dim3(gM, MLPD_ / 128), blk, 0, stream>>>(
            xnh, wh, fc1_b + (size_t)i * MLPD_,
            nullptr, nullptr, mlph, M_ROWS, MLPD_, D_, 1);
        // FC2 GEMM + residual (6304 x 768 x 3072)
        cvt(fc2_w + (size_t)i * D_ * MLPD_, D_ * MLPD_);
        gemm_f16_wmma<<<dim3(gM, D_ / 128), blk, 0, stream>>>(
            mlph, wh, fc2_b + (size_t)i * D_,
            hbuf, hbuf, nullptr, M_ROWS, D_, MLPD_, 0);
    }

    // final LN on CLS rows only -> d_out (32 x 768 f32)
    layernorm_kernel<<<dim3(B_), blk, 0, stream>>>(
        hbuf, norm_w, norm_b, nullptr, (float*)d_out, N_);
}
